// GAT_75677323755528
// MI455X (gfx1250) — compile-verified, matrix operand order checked by reading, not run on hardware
//
#include <hip/hip_runtime.h>
#include <math.h>

#define NEG_SLOPE 0.2f
#define K_DIM 128      // both layers have K = 128
#define KTILES (K_DIM / 32)

typedef __attribute__((ext_vector_type(16))) _Float16 v16h;
typedef __attribute__((ext_vector_type(8)))  _Float16 v8h;
typedef __attribute__((ext_vector_type(8)))  float    v8f;

// ------------------------------------------------------------------
// Elementwise helpers
// ------------------------------------------------------------------
__global__ void k_f32_to_f16(const float* __restrict__ in, _Float16* __restrict__ out, int n) {
  int i = blockIdx.x * blockDim.x + threadIdx.x;
  if (i < n) out[i] = (_Float16)in[i];
}

__global__ void k_fill_f32(float* __restrict__ p, float v, int n) {
  int i = blockIdx.x * blockDim.x + threadIdx.x;
  if (i < n) p[i] = v;
}

// ------------------------------------------------------------------
// Pack W [K x Nout] (f32, row-major) into WMMA B-fragment order (f16).
// Fragment (kt, nt): 32 lanes x 16 halves.  Lane c (c<16) = column nt*16+c,
// holds K = kt*32 + {0..7, 16..23}; lane c+16 holds K = kt*32 + {8..15, 24..31}.
// ------------------------------------------------------------------
__global__ void k_pack_w(const float* __restrict__ W, _Float16* __restrict__ Bp,
                         int K, int Nout) {
  int tid = blockIdx.x * blockDim.x + threadIdx.x;
  int ntn = Nout >> 4;
  int total = (K >> 5) * ntn * 32;
  if (tid >= total) return;
  int lane = tid & 31;
  int frag = tid >> 5;
  int kt = frag / ntn;
  int nt = frag - kt * ntn;
  int col  = (nt << 4) + (lane & 15);
  int hsel = lane >> 4;
  _Float16* out = Bp + (size_t)tid * 16;
#pragma unroll
  for (int j = 0; j < 8; ++j) {
    int k0 = (kt << 5) + (hsel << 3) + j;
    out[j]     = (_Float16)W[(size_t)k0 * Nout + col];
    out[j + 8] = (_Float16)W[(size_t)(k0 + 16) * Nout + col];
  }
}

// ------------------------------------------------------------------
// C[M x NOUT] (f32) = A[M x 128] (f16, row-major) * Bpacked (f16 fragments).
// Whole packed B (<=32KB) is staged in LDS once per workgroup; each wave then
// computes a full 16 x NOUT strip: A fragments loaded once, KTILES*NOUT/16
// v_wmma_f32_16x16x32_f16 issued per wave, B fed by ds_load_b128.
// ------------------------------------------------------------------
template <int NOUT>
__global__ void k_wmma_gemm(const _Float16* __restrict__ A,
                            const _Float16* __restrict__ Bp,
                            float* __restrict__ C, int M) {
  constexpr int NTN = NOUT / 16;
  constexpr int BHALVES = KTILES * NTN * 32 * 16;       // 16K halves @128, 8K @64
  __shared__ __align__(32) _Float16 sB[BHALVES];

  // cooperative stage of packed B into LDS (16B chunks)
  for (int idx = threadIdx.x; idx < BHALVES / 8; idx += blockDim.x)
    *(v8h*)(sB + (size_t)idx * 8) = *(const v8h*)(Bp + (size_t)idx * 8);
  __syncthreads();

  const int lane = threadIdx.x & 31;
  const int mt = blockIdx.x * (blockDim.x >> 5) + (threadIdx.x >> 5);
  if (mt >= (M >> 4)) return;   // uniform per wave; after the barrier

  const int m    = lane & 15;
  const int hsel = lane >> 4;
  const int row  = (mt << 4) + m;

  v8f acc[NTN] = {};
#pragma unroll
  for (int kt = 0; kt < KTILES; ++kt) {
    // A fragment: lane m holds K {kt*32 + hsel*8 .. +7} and {.. +16 ..}
    const int abase = row * K_DIM + (kt << 5) + (hsel << 3);
    v8h a0 = *(const v8h*)(A + abase);
    v8h a1 = *(const v8h*)(A + abase + 16);
    v16h a;
#pragma unroll
    for (int j = 0; j < 8; ++j) { a[j] = a0[j]; a[j + 8] = a1[j]; }
#pragma unroll
    for (int nt = 0; nt < NTN; ++nt) {
      const v16h b = *(const v16h*)(sB + ((size_t)(kt * NTN + nt) * 32 + lane) * 16);
      acc[nt] = __builtin_amdgcn_wmma_f32_16x16x32_f16(false, a, false, b,
                                                       (short)0, acc[nt], false, false);
    }
  }
  // C/D layout: lanes 0-15 -> N=lane, M=v; lanes 16-31 -> N=lane-16, M=8+v
  const int row0 = (mt << 4) + (hsel << 3);
#pragma unroll
  for (int nt = 0; nt < NTN; ++nt) {
    const int col = (nt << 4) + (lane & 15);
#pragma unroll
    for (int v = 0; v < 8; ++v)
      C[(size_t)(row0 + v) * NOUT + col] = acc[nt][v];
  }
}

// ------------------------------------------------------------------
// Per-node attention scores: a_src[n] = <h_src[n,:], att_src>, same for dst.
// One wave per node, shuffle reduction (wave32).
// ------------------------------------------------------------------
__global__ void k_attn_scores(const float* __restrict__ hsrc, const float* __restrict__ hdst,
                              const float* __restrict__ att_src, const float* __restrict__ att_dst,
                              float* __restrict__ a_src, float* __restrict__ a_dst,
                              int n, int C) {
  int lane = threadIdx.x & 31;
  int node = blockIdx.x * (blockDim.x >> 5) + (threadIdx.x >> 5);
  if (node >= n) return;
  float s = 0.f, d = 0.f;
  for (int c = lane; c < C; c += 32) {
    s += hsrc[(size_t)node * C + c] * att_src[c];
    d += hdst[(size_t)node * C + c] * att_dst[c];
  }
#pragma unroll
  for (int o = 16; o > 0; o >>= 1) {
    s += __shfl_down(s, o, 32);
    d += __shfl_down(d, o, 32);
  }
  if (lane == 0) { a_src[node] = s; a_dst[node] = d; }
}

// ------------------------------------------------------------------
// Edge phase
// ------------------------------------------------------------------
__device__ __forceinline__ void atomicMaxF(float* addr, float v) {
  // sign-aware max on the raw bits; correct for all finite f32 and -inf init
  if (v >= 0.f) atomicMax((int*)addr, __float_as_int(v));
  else          atomicMin((unsigned int*)addr, __float_as_uint(v));
}

__global__ void k_edge_logits(const int* __restrict__ src, const int* __restrict__ dst,
                              const float* __restrict__ a_src, const float* __restrict__ a_dst,
                              float* __restrict__ e, float* __restrict__ m, int E) {
  int i = blockIdx.x * blockDim.x + threadIdx.x;
  if (i >= E) return;
  float v = a_src[src[i]] + a_dst[dst[i]];
  v = v > 0.f ? v : NEG_SLOPE * v;
  e[i] = v;
  atomicMaxF(&m[dst[i]], v);
}

__global__ void k_edge_exp(const int* __restrict__ dst, const float* __restrict__ m,
                           float* __restrict__ e /* in: logits, out: p */,
                           float* __restrict__ denom, int E) {
  int i = blockIdx.x * blockDim.x + threadIdx.x;
  if (i >= E) return;
  int d = dst[i];
  float p = __expf(e[i] - m[d]);   // m[d] finite: node d has >=1 incoming edge
  e[i] = p;
  atomicAdd(&denom[d], p);
}

// One wave per edge: conv[dst,:] += h_src[src,:] * alpha   (coalesced gather +
// global_atomic_add_f32 scatter)
__global__ void k_edge_scatter(const int* __restrict__ src, const int* __restrict__ dst,
                               const float* __restrict__ p, const float* __restrict__ denom,
                               const float* __restrict__ hsrc, float* __restrict__ conv,
                               int E, int C) {
  int lane = threadIdx.x & 31;
  int ei = blockIdx.x * (blockDim.x >> 5) + (threadIdx.x >> 5);
  if (ei >= E) return;
  int s = src[ei], d = dst[ei];
  float alpha = p[ei] / (denom[d] + 1e-16f);
  for (int c = lane; c < C; c += 32)
    atomicAdd(&conv[(size_t)d * C + c], hsrc[(size_t)s * C + c] * alpha);
}

// out = conv + b_conv + lin + b_lin  (optional ReLU)
__global__ void k_combine(const float* __restrict__ conv, const float* __restrict__ bconv,
                          const float* __restrict__ lin, const float* __restrict__ blin,
                          float* __restrict__ out, int n, int C, int do_relu) {
  int i = blockIdx.x * blockDim.x + threadIdx.x;
  if (i >= n * C) return;
  int c = i % C;
  float v = conv[i] + bconv[c] + lin[i] + blin[c];
  out[i] = do_relu ? fmaxf(v, 0.f) : v;
}

// ------------------------------------------------------------------
// Host orchestration
// ------------------------------------------------------------------
extern "C" void kernel_launch(void* const* d_in, const int* in_sizes, int n_in,
                              void* d_out, int out_size, void* d_ws, size_t ws_size,
                              hipStream_t stream) {
  const float* x        = (const float*)d_in[0];
  const int*   eidx     = (const int*)d_in[1];   // int32 (jax x64 off)
  const float* Wsrc1    = (const float*)d_in[2];
  const float* Wdst1    = (const float*)d_in[3];
  const float* att_src1 = (const float*)d_in[4];
  const float* att_dst1 = (const float*)d_in[5];
  const float* b_conv1  = (const float*)d_in[6];
  const float* Wlin1    = (const float*)d_in[7];
  const float* b_lin1   = (const float*)d_in[8];
  const float* Wsrc2    = (const float*)d_in[9];
  const float* Wdst2    = (const float*)d_in[10];
  const float* att_src2 = (const float*)d_in[11];
  const float* att_dst2 = (const float*)d_in[12];
  const float* b_conv2  = (const float*)d_in[13];
  const float* Wlin2    = (const float*)d_in[14];
  const float* b_lin2   = (const float*)d_in[15];

  const int N   = in_sizes[0] / 128;   // 50000 (multiple of 16)
  const int E   = in_sizes[1] / 2;     // 800000
  const int HID = 128, OC = 64;
  const int* src = eidx;
  const int* dst = eidx + E;

  // ---- workspace carve (256B aligned), with reuse across phases ----
  auto al = [](size_t v) { return (v + 255) & ~(size_t)255; };
  char* ws = (char*)d_ws;
  size_t off = 0;
  _Float16* A16  = (_Float16*)(ws + off); off += al((size_t)N * 128 * sizeof(_Float16));
  float* B1      = (float*)(ws + off);    off += al((size_t)N * 128 * sizeof(float)); // hsrc1 -> h_out
  float* B2      = (float*)(ws + off);    off += al((size_t)N * 128 * sizeof(float)); // hdst1 -> conv1
  float* B3      = (float*)(ws + off);    off += al((size_t)N * 128 * sizeof(float)); // lin1 -> hsrc2|hdst2
  float* lin2    = (float*)(ws + off);    off += al((size_t)N * 64 * sizeof(float));
  float* conv2   = (float*)(ws + off);    off += al((size_t)N * 64 * sizeof(float));
  float* a_srcB  = (float*)(ws + off);    off += al((size_t)N * sizeof(float));
  float* a_dstB  = (float*)(ws + off);    off += al((size_t)N * sizeof(float));
  float* mB      = (float*)(ws + off);    off += al((size_t)N * sizeof(float));
  float* denomB  = (float*)(ws + off);    off += al((size_t)N * sizeof(float));
  float* ebuf    = (float*)(ws + off);    off += al((size_t)E * sizeof(float));
  _Float16* Wp   = (_Float16*)(ws + off); off += al((size_t)128 * 128 * sizeof(_Float16));
  if (off > ws_size) return;  // insufficient scratch

  float* hsrc2 = B3;
  float* hdst2 = B3 + (size_t)N * 64;

  const int B256 = 256, B128 = 128;
  auto cdiv = [](int a, int b) { return (a + b - 1) / b; };

  // =============== Layer 1 ===============
  k_f32_to_f16<<<cdiv(N * 128, B256), B256, 0, stream>>>(x, A16, N * 128);

  int packW1 = cdiv((128 / 32) * (HID / 16) * 32, B256);
  int gemmB  = cdiv(N / 16, 4);     // one 16-row strip per wave, 4 waves/block
  k_pack_w<<<packW1, B256, 0, stream>>>(Wsrc1, Wp, 128, HID);
  k_wmma_gemm<128><<<gemmB, B128, 0, stream>>>(A16, Wp, B1, N);
  k_pack_w<<<packW1, B256, 0, stream>>>(Wdst1, Wp, 128, HID);
  k_wmma_gemm<128><<<gemmB, B128, 0, stream>>>(A16, Wp, B2, N);
  k_pack_w<<<packW1, B256, 0, stream>>>(Wlin1, Wp, 128, HID);
  k_wmma_gemm<128><<<gemmB, B128, 0, stream>>>(A16, Wp, B3, N);

  k_attn_scores<<<cdiv(N, 4), B128, 0, stream>>>(B1, B2, att_src1, att_dst1,
                                                 a_srcB, a_dstB, N, HID);
  k_fill_f32<<<cdiv(N, B256), B256, 0, stream>>>(mB, -INFINITY, N);
  k_fill_f32<<<cdiv(N, B256), B256, 0, stream>>>(denomB, 0.f, N);
  k_edge_logits<<<cdiv(E, B256), B256, 0, stream>>>(src, dst, a_srcB, a_dstB, ebuf, mB, E);
  k_edge_exp<<<cdiv(E, B256), B256, 0, stream>>>(dst, mB, ebuf, denomB, E);
  k_fill_f32<<<cdiv(N * 128, B256), B256, 0, stream>>>(B2, 0.f, N * 128);  // conv1 accum
  k_edge_scatter<<<cdiv(E, 4), B128, 0, stream>>>(src, dst, ebuf, denomB, B1, B2, E, HID);
  // h = relu(conv1 + b_conv1 + lin1 + b_lin1)  -> B1 (hsrc1 no longer needed)
  k_combine<<<cdiv(N * 128, B256), B256, 0, stream>>>(B2, b_conv1, B3, b_lin1, B1, N, HID, 1);

  // =============== Layer 2 ===============
  k_f32_to_f16<<<cdiv(N * 128, B256), B256, 0, stream>>>(B1, A16, N * 128);

  int packW2 = cdiv((128 / 32) * (OC / 16) * 32, B256);
  k_pack_w<<<packW2, B256, 0, stream>>>(Wsrc2, Wp, 128, OC);
  k_wmma_gemm<64><<<gemmB, B128, 0, stream>>>(A16, Wp, hsrc2, N);
  k_pack_w<<<packW2, B256, 0, stream>>>(Wdst2, Wp, 128, OC);
  k_wmma_gemm<64><<<gemmB, B128, 0, stream>>>(A16, Wp, hdst2, N);
  k_pack_w<<<packW2, B256, 0, stream>>>(Wlin2, Wp, 128, OC);
  k_wmma_gemm<64><<<gemmB, B128, 0, stream>>>(A16, Wp, lin2, N);

  k_attn_scores<<<cdiv(N, 4), B128, 0, stream>>>(hsrc2, hdst2, att_src2, att_dst2,
                                                 a_srcB, a_dstB, N, OC);
  k_fill_f32<<<cdiv(N, B256), B256, 0, stream>>>(mB, -INFINITY, N);
  k_fill_f32<<<cdiv(N, B256), B256, 0, stream>>>(denomB, 0.f, N);
  k_edge_logits<<<cdiv(E, B256), B256, 0, stream>>>(src, dst, a_srcB, a_dstB, ebuf, mB, E);
  k_edge_exp<<<cdiv(E, B256), B256, 0, stream>>>(dst, mB, ebuf, denomB, E);
  k_fill_f32<<<cdiv(N * 64, B256), B256, 0, stream>>>(conv2, 0.f, N * 64);
  k_edge_scatter<<<cdiv(E, 4), B128, 0, stream>>>(src, dst, ebuf, denomB, hsrc2, conv2, E, OC);
  k_combine<<<cdiv(N * 64, B256), B256, 0, stream>>>(conv2, b_conv2, lin2, b_lin2,
                                                     (float*)d_out, N, OC, 0);
}
